// CleanDifferentiableTensorSketch_21818433864269
// MI455X (gfx1250) — matrix operand order repaired and checked by reading, not compiled, not run on hardware
//
#include <hip/hip_runtime.h>
#include <math.h>

// CleanDifferentiableTensorSketch — frequency-domain scan implementation.
// L=8 levels, A=4 symbols, D=256, SEQ=16384.
//
// Pipeline:
//  K1: softmax over hash_emb -> probs[32][256], sigmoid(sign_logits) -> S[32]
//  K2: DFT of the 32 prob vectors via split-f16 WMMA GEMM (fp32-class accuracy)
//  K3: per-frequency scan over 16384 steps (256 independent lanes, 8 waves)
//  K4: inverse DFT of (u_p[L]-u_m[L]) -> 256 real outputs

#define LV 8
#define NA 4
#define DD 256
#define NSEQ 16384

typedef __attribute__((ext_vector_type(16))) _Float16 v16h;
typedef __attribute__((ext_vector_type(8)))  float    v8f;

// ---------------- Kernel 1: softmax + sigmoid ----------------
__global__ void k_softmax(const float* __restrict__ hash_emb,
                          const float* __restrict__ sign_logits,
                          float* __restrict__ probs,
                          float* __restrict__ S) {
    __shared__ float red[256];
    const int lc = blockIdx.x;      // lc = l*4 + c, 0..31
    const int t  = threadIdx.x;     // 0..255
    float x = hash_emb[lc * DD + t];
    red[t] = x;
    __syncthreads();
    for (int s = 128; s > 0; s >>= 1) {
        if (t < s) red[t] = fmaxf(red[t], red[t + s]);
        __syncthreads();
    }
    const float mx = red[0];
    __syncthreads();
    const float e = expf(x - mx);
    red[t] = e;
    __syncthreads();
    for (int s = 128; s > 0; s >>= 1) {
        if (t < s) red[t] = red[t] + red[t + s];
        __syncthreads();
    }
    probs[lc * DD + t] = e / red[0];
    if (t == 0) S[lc] = 1.0f / (1.0f + expf(-sign_logits[lc]));
}

// ---------------- Kernel 2: DFT of probs via split-f16 WMMA ----------------
// Grid: 32 blocks of 32 threads (1 wave each). blockIdx = mt + 2*nt.
// C tile = probs[mt*16 .. +16][:] @ W[:, nt*16 .. +16], K = 256 in 8 chunks of 32.
// Twiddles via __sincosf: args pre-reduced to [0, 2*pi) in integers, so the
// hardware sin/cos path is branch-free and accurate (~1e-6 abs).
__global__ void k_dft_wmma(const float* __restrict__ probs,
                           float* __restrict__ Pre,
                           float* __restrict__ Pim) {
    const int lane = threadIdx.x;
    const int half = lane >> 4;      // 0 or 1
    const int mn   = lane & 15;
    const int mt   = blockIdx.x & 1;
    const int nt   = blockIdx.x >> 1;
    const int arow  = mt * 16 + mn;  // A-matrix row this lane loads
    const int kfreq = nt * 16 + mn;  // output frequency column
    const float w = 6.28318530717958647692f / 256.0f;

    v8f acc_c = {0.f,0.f,0.f,0.f,0.f,0.f,0.f,0.f};
    v8f acc_s = {0.f,0.f,0.f,0.f,0.f,0.f,0.f,0.f};

#pragma unroll
    for (int kb = 0; kb < 8; ++kb) {
        const int sbase = kb * 32;
        v16h Ah, Al, Bhc, Blc, Bhs, Bls;
#pragma unroll
        for (int j = 0; j < 16; ++j) {
            // 16-bit A 16x32 layout: lanes0-15 K = (j<8 ? j : j+8),
            // lanes16-31 add 8 (ISA 7.12.2).
            const int Ka = (j < 8 ? j : j + 8) + (half ? 8 : 0);
            const float x = probs[arow * DD + sbase + Ka];
            const _Float16 xh = (_Float16)x;
            Ah[j] = xh;
            Al[j] = (_Float16)(x - (float)xh);
            // B 32x16: same K order as A (lanes16-31 add 8).
            const int Kb = (j < 8 ? j : j + 8) + (half ? 8 : 0);
            const int tt = ((sbase + Kb) * kfreq) & 255;
            float sn, cs;
            __sincosf((float)tt * w, &sn, &cs);
            const _Float16 ch = (_Float16)cs;
            Bhc[j] = ch; Blc[j] = (_Float16)(cs - (float)ch);
            const _Float16 sh = (_Float16)sn;
            Bhs[j] = sh; Bls[j] = (_Float16)(sn - (float)sh);
        }
        // Split product: x*y ~= xh*yh + xh*yl + xl*yh (f32 accumulation)
        acc_c = __builtin_amdgcn_wmma_f32_16x16x32_f16(false, Ah, false, Bhc, (short)0, acc_c, false, false);
        acc_c = __builtin_amdgcn_wmma_f32_16x16x32_f16(false, Ah, false, Blc, (short)0, acc_c, false, false);
        acc_c = __builtin_amdgcn_wmma_f32_16x16x32_f16(false, Al, false, Bhc, (short)0, acc_c, false, false);
        acc_s = __builtin_amdgcn_wmma_f32_16x16x32_f16(false, Ah, false, Bhs, (short)0, acc_s, false, false);
        acc_s = __builtin_amdgcn_wmma_f32_16x16x32_f16(false, Ah, false, Bls, (short)0, acc_s, false, false);
        acc_s = __builtin_amdgcn_wmma_f32_16x16x32_f16(false, Al, false, Bhs, (short)0, acc_s, false, false);
    }
#pragma unroll
    for (int r = 0; r < 8; ++r) {
        // C/D 16x16 f32 layout: VGPR r, lanes0-15 -> M=r, lanes16-31 -> M=8+r
        const int row = mt * 16 + half * 8 + r;
        Pre[row * DD + kfreq] =  acc_c[r];
        Pim[row * DD + kfreq] = -acc_s[r];   // e^{-i theta} convention
    }
}

// ---------------- Kernel 3: per-frequency scan ----------------
// One complex-valued chain per lane (frequency k). State fully in registers;
// per step: 1 scalar load (c), v_rcp, 8x ds_load_b128, ~22 VALU per level.

template <bool EARLY>
__device__ __forceinline__ void scan_step(int i, int c, int lane,
                                          const float4* __restrict__ tab4,
                                          float (&upr)[LV + 1], float (&upi)[LV + 1],
                                          float (&umr)[LV + 1], float (&umi)[LV + 1]) {
    const float inv = __builtin_amdgcn_rcpf((float)(i + 1));  // ~1 ulp, fine for z
    const int base = c * 32 + lane;
    // Descending levels: lo reads old level lo-1, writes level lo in place.
#pragma unroll
    for (int lo = LV; lo >= 1; --lo) {
        const int l = lo - 1;
        const float4 e = tab4[base + l * 128];  // (Pre, Pim, s, 0) for (l,c,k)
        const float Pr = e.x, Pi = e.y, s = e.z;
        float z = (float)lo * inv;
        if (EARLY) z = (i >= l) ? z : 0.0f;
        const float ar = upr[l], ai = upi[l], br = umr[l], bi = umi[l];
        // stp = u_p[l] * P ; stm = u_m[l] * P  (complex)
        const float stpr = ar * Pr - ai * Pi;
        const float stpi = ar * Pi + ai * Pr;
        const float stmr = br * Pr - bi * Pi;
        const float stmi = br * Pi + bi * Pr;
        const float tr = stpr - stmr, ti = stpi - stmi;
        const float updpr = fmaf(s, tr, stmr);
        const float updpi = fmaf(s, ti, stmi);
        const float updmr = fmaf(-s, tr, stpr);
        const float updmi = fmaf(-s, ti, stpi);
        upr[lo] = fmaf(z, updpr - upr[lo], upr[lo]);
        upi[lo] = fmaf(z, updpi - upi[lo], upi[lo]);
        umr[lo] = fmaf(z, updmr - umr[lo], umr[lo]);
        umi[lo] = fmaf(z, updmi - umi[lo], umi[lo]);
    }
}

__global__ void k_scan(const int* __restrict__ seq,
                       const float* __restrict__ Pre,
                       const float* __restrict__ Pim,
                       const float* __restrict__ S,
                       float* __restrict__ Ure,
                       float* __restrict__ Uim) {
    __shared__ float4 tab4[LV * NA * 32];   // 16 KB
    const int lane = threadIdx.x;
    const int k = blockIdx.x * 32 + lane;

    for (int lc = 0; lc < LV * NA; ++lc) {
        float4 v;
        v.x = Pre[lc * DD + k];
        v.y = Pim[lc * DD + k];
        v.z = S[lc];
        v.w = 0.0f;
        tab4[lc * 32 + lane] = v;
    }
    __syncthreads();

    // Complex state per level: u_p, u_m. Level 0 is constant (1,0)/(0,0).
    float upr[LV + 1], upi[LV + 1], umr[LV + 1], umi[LV + 1];
#pragma unroll
    for (int l = 0; l <= LV; ++l) { upr[l] = 0.f; upi[l] = 0.f; umr[l] = 0.f; umi[l] = 0.f; }
    upr[0] = 1.0f;   // FFT of delta at index 0

    // Peeled prologue: steps 0..7 need the (p <= i+1) mask.
    for (int i = 0; i < LV; ++i) {
        scan_step<true>(i, seq[i], lane, tab4, upr, upi, umr, umi);
    }
    // Hot loop: mask always true.
#pragma unroll 2
    for (int i = LV; i < NSEQ; ++i) {
        scan_step<false>(i, seq[i], lane, tab4, upr, upi, umr, umi);
    }

    Ure[k] = upr[LV] - umr[LV];
    Uim[k] = upi[LV] - umi[LV];
}

// ---------------- Kernel 4: inverse DFT ----------------
__global__ void k_idft(const float* __restrict__ Ure,
                       const float* __restrict__ Uim,
                       float* __restrict__ out) {
    __shared__ float ur[DD], ui[DD], ct[DD], st[DD];
    const int j = threadIdx.x;
    const float w = 6.28318530717958647692f / 256.0f;
    ur[j] = Ure[j];
    ui[j] = Uim[j];
    float sn, cs;
    sincosf((float)j * w, &sn, &cs);
    ct[j] = cs; st[j] = sn;
    __syncthreads();
    float acc = 0.0f;
    for (int k = 0; k < DD; ++k) {
        const int t = (j * k) & 255;
        acc = fmaf(ur[k], ct[t], acc);
        acc = fmaf(-ui[k], st[t], acc);
    }
    out[j] = acc * (1.0f / 256.0f);
}

// ---------------- Launch ----------------
extern "C" void kernel_launch(void* const* d_in, const int* in_sizes, int n_in,
                              void* d_out, int out_size, void* d_ws, size_t ws_size,
                              hipStream_t stream) {
    const int*   seq         = (const int*)d_in[0];    // (16384,) int32
    const float* hash_emb    = (const float*)d_in[1];  // (8,4,256) f32
    const float* sign_logits = (const float*)d_in[2];  // (8,4) f32
    float* out = (float*)d_out;                        // (256,) f32
    float* ws  = (float*)d_ws;

    float* probs = ws;            // 8192 floats
    float* Pre   = ws + 8192;     // 8192
    float* Pim   = ws + 16384;    // 8192
    float* S     = ws + 24576;    // 32
    float* Ure   = ws + 24640;    // 256
    float* Uim   = ws + 24896;    // 256

    k_softmax <<<32, 256, 0, stream>>>(hash_emb, sign_logits, probs, S);
    k_dft_wmma<<<32,  32, 0, stream>>>(probs, Pre, Pim);
    k_scan    <<< 8,  32, 0, stream>>>(seq, Pre, Pim, S, Ure, Uim);
    k_idft    <<< 1, 256, 0, stream>>>(Ure, Uim, out);
}
